// _NonLocalBlockND_8478265442595
// MI455X (gfx1250) — compile-verified, matrix operand order checked by reading, not run on hardware
//
#include <hip/hip_runtime.h>
#include <hip/hip_bf16.h>
#include <cstdint>

typedef _Float16 f16;
typedef __attribute__((ext_vector_type(16))) _Float16 v16h;
typedef __attribute__((ext_vector_type(8)))  float    v8f;
typedef __attribute__((ext_vector_type(4)))  unsigned int u32x4;
typedef __attribute__((ext_vector_type(8)))  int i32x8;
typedef __attribute__((ext_vector_type(4)))  int i32x4;

#define B_  16
#define C_  256
#define CI_ 128
#define N_  4096   // 64*64
#define M_  1024   // 32*32
#define HW_ 64

#if defined(__HIP_DEVICE_COMPILE__) && __has_builtin(__builtin_amdgcn_tensor_load_to_lds)
#define HAVE_TDM 1
#else
#define HAVE_TDM 0
#endif

// workspace byte offsets (all 256-aligned)
static constexpr size_t OFF_WBF  = 0;                       // f16 weights: theta|phi|g (each 128*256) then W (256*128)
static constexpr size_t OFF_TH   = 262144;                  // f16 [B][128][4096]
static constexpr size_t OFF_PHIF = OFF_TH   + 16777216;     // f16 [B][128][4096]
static constexpr size_t OFF_GF   = OFF_PHIF + 16777216;     // f16 [B][128][4096]
static constexpr size_t OFF_P    = OFF_GF   + 16777216;     // f16 [B][128][1024]  phi pooled
static constexpr size_t OFF_GT   = OFF_P    + 4194304;      // f16 [B][1024][128]  g pooled, transposed
static constexpr size_t OFF_T    = OFF_GT   + 4194304;      // f16 [B][256][128]   T = W @ S^T
static constexpr size_t OFF_SUM  = OFF_T    + 1048576;      // f32 [256]
static constexpr size_t OFF_SSQ  = OFF_SUM  + 1024;         // f32 [256]

// ---------------- WMMA helpers (gfx1250, wave32) ----------------

__device__ __forceinline__ v8f wmma16(v16h a, v16h b, v8f c) {
  // v_wmma_f32_16x16x32_f16: D = A(16x32) * B(32x16) + C(16x16 f32)
  return __builtin_amdgcn_wmma_f32_16x16x32_f16(false, a, false, b, (short)0, c, false, false);
}

// A fragment (16x32, f16) from row-major matrix: lane l<16 -> row l, K in {0..7,16..23};
// lane l+16 -> row l, K in {8..15,24..31} (ISA 16-bit A layout).
__device__ __forceinline__ v16h load_a16(const f16* __restrict__ A, int lda, int row0, int k0, int lane) {
  const int l = lane & 15, hi = lane >> 4;
  const f16* p = A + (size_t)(row0 + l) * lda + k0 + hi * 8;
  v16h r;
#pragma unroll
  for (int e = 0; e < 8; ++e) { r[e] = p[e]; r[e + 8] = p[16 + e]; }
  return r;
}

// B fragment (32x16, f16) from row-major K x N matrix: lane holds one column, same K split.
__device__ __forceinline__ v16h load_b16(const f16* __restrict__ Bm, int ldb, int k0, int col0, int lane) {
  const int l = lane & 15, hi = lane >> 4;
  const f16* p = Bm + (size_t)(k0 + hi * 8) * ldb + col0 + l;
  v16h r;
#pragma unroll
  for (int e = 0; e < 8; ++e) { r[e] = p[(size_t)e * ldb]; r[e + 8] = p[(size_t)(16 + e) * ldb]; }
  return r;
}

// B fragment loaded from f32 source with on-the-fly f16 convert (for x).
__device__ __forceinline__ v16h load_b32f(const float* __restrict__ Bm, int ldb, int k0, int col0, int lane) {
  const int l = lane & 15, hi = lane >> 4;
  const float* p = Bm + (size_t)(k0 + hi * 8) * ldb + col0 + l;
  v16h r;
#pragma unroll
  for (int e = 0; e < 8; ++e) { r[e] = (f16)p[(size_t)e * ldb]; r[e + 8] = (f16)p[(size_t)(16 + e) * ldb]; }
  return r;
}

// ---------------- kernels ----------------

// init: zero BN accumulators, convert all weights f32 -> f16
__global__ void k0_init(const float* __restrict__ tw, const float* __restrict__ pw,
                        const float* __restrict__ gw, const float* __restrict__ Ww,
                        f16* __restrict__ wbf, float* __restrict__ gsum, float* __restrict__ gssq) {
  int idx = blockIdx.x * 256 + threadIdx.x;
  if (idx < 512) { if (idx < 256) gsum[idx] = 0.f; else gssq[idx - 256] = 0.f; }
  if (idx < 32768)        wbf[idx] = (f16)tw[idx];
  else if (idx < 65536)   wbf[idx] = (f16)pw[idx - 32768];
  else if (idx < 98304)   wbf[idx] = (f16)gw[idx - 65536];
  else if (idx < 131072)  wbf[idx] = (f16)Ww[idx - 98304];
}

// conv1x1: out[b][ci][n] = W[ci][:] . x[b][:][n] + bias[ci], stored f16 at full res.
// grid (N/128, CI/64, B), block 256 (8 waves, 2 row-groups x 4 col-groups, 32x32 per wave)
__global__ void __launch_bounds__(256) k1_conv(const float* __restrict__ x, const f16* __restrict__ Wt,
                                               const float* __restrict__ bias, f16* __restrict__ out) {
  const int lane = threadIdx.x & 31, wid = threadIdx.x >> 5;
  const int wr = wid & 1, wc = wid >> 1;
  const int b  = blockIdx.z;
  const int r0 = blockIdx.y * 64 + wr * 32;
  const int c0 = blockIdx.x * 128 + wc * 32;
  const float* xb = x + (size_t)b * C_ * N_;
  f16* ob = out + (size_t)b * CI_ * N_;
  v8f acc[2][2] = {};
  for (int k0 = 0; k0 < C_; k0 += 32) {
    if (k0 + 32 < C_) __builtin_prefetch(xb + (size_t)(k0 + 32) * N_ + c0 + (lane & 15), 0, 3);
    v16h a0 = load_a16(Wt, C_, r0,      k0, lane);
    v16h a1 = load_a16(Wt, C_, r0 + 16, k0, lane);
    v16h b0 = load_b32f(xb, N_, k0, c0,      lane);
    v16h b1 = load_b32f(xb, N_, k0, c0 + 16, lane);
    acc[0][0] = wmma16(a0, b0, acc[0][0]);
    acc[0][1] = wmma16(a0, b1, acc[0][1]);
    acc[1][0] = wmma16(a1, b0, acc[1][0]);
    acc[1][1] = wmma16(a1, b1, acc[1][1]);
  }
  const int l = lane & 15, hi = lane >> 4;
#pragma unroll
  for (int i = 0; i < 2; ++i) {
#pragma unroll
    for (int r = 0; r < 8; ++r) {
      int row = r0 + 16 * i + hi * 8 + r;
      float wb = bias[row];
#pragma unroll
      for (int j = 0; j < 2; ++j) {
        int col = c0 + 16 * j + l;
        ob[(size_t)row * N_ + col] = (f16)(acc[i][j][r] + wb);
      }
    }
  }
}

// 2x2 max pool; optional transposed write (for g -> Gt[b][m][ci])
__global__ void k2_pool(const f16* __restrict__ src, f16* __restrict__ dst, int transpose) {
  int idx = blockIdx.x * 256 + threadIdx.x;      // B*CI*M
  int m  = idx & (M_ - 1);
  int ci = (idx >> 10) & (CI_ - 1);
  int b  = idx >> 17;
  int mh = m >> 5, mw = m & 31;
  const f16* s = src + (size_t)b * CI_ * N_ + (size_t)ci * N_ + (size_t)(2 * mh) * HW_ + 2 * mw;
  float v0 = (float)s[0], v1 = (float)s[1], v2 = (float)s[HW_], v3 = (float)s[HW_ + 1];
  float mx = fmaxf(fmaxf(v0, v1), fmaxf(v2, v3));
  if (transpose) dst[(size_t)b * M_ * CI_ + (size_t)m * CI_ + ci] = (f16)mx;
  else           dst[(size_t)b * CI_ * M_ + (size_t)ci * M_ + m]  = (f16)mx;
}

// per-batch: S = (P @ Gt)/M  (128x128, staged in LDS), then T = W @ S^T (256x128) -> f16
__global__ void __launch_bounds__(256) k3_ST(const f16* __restrict__ P, const f16* __restrict__ Gt,
                                             const f16* __restrict__ Wbf, f16* __restrict__ Tout) {
  __shared__ f16 Slds[128 * 128];
  const int lane = threadIdx.x & 31, wid = threadIdx.x >> 5;
  const int l = lane & 15, hi = lane >> 4;
  const int b = blockIdx.x;
  const f16* Pb = P  + (size_t)b * CI_ * M_;
  const f16* Gb = Gt + (size_t)b * M_ * CI_;
  { // phase 1: rows 16*wid .. +15 of S, all 128 cols
    v8f acc[8] = {};
    const int r0 = 16 * wid;
    for (int k0 = 0; k0 < M_; k0 += 32) {
      v16h a = load_a16(Pb, M_, r0, k0, lane);
#pragma unroll
      for (int t = 0; t < 8; ++t) {
        v16h bt = load_b16(Gb, CI_, k0, 16 * t, lane);
        acc[t] = wmma16(a, bt, acc[t]);
      }
    }
#pragma unroll
    for (int t = 0; t < 8; ++t)
#pragma unroll
      for (int r = 0; r < 8; ++r) {
        int row = r0 + hi * 8 + r, col = 16 * t + l;
        Slds[row * 128 + col] = (f16)(acc[t][r] * (1.0f / (float)M_));
      }
  }
  __syncthreads();
  { // phase 2: T[c][i] = sum_j W[c][j] * S[i][j]  -> B operand is rows of S (A-pattern load)
    const int r0 = 32 * wid;
#pragma unroll 1
    for (int cj = 0; cj < 8; ++cj) {
      v8f acc[2] = {};
#pragma unroll
      for (int kk = 0; kk < 4; ++kk) {
        int k0 = kk * 32;
        v16h a0 = load_a16(Wbf, CI_, r0,      k0, lane);
        v16h a1 = load_a16(Wbf, CI_, r0 + 16, k0, lane);
        v16h bb = load_a16(Slds, 128, 16 * cj, k0, lane);  // B[k][col] = S[col][k]
        acc[0] = wmma16(a0, bb, acc[0]);
        acc[1] = wmma16(a1, bb, acc[1]);
      }
#pragma unroll
      for (int i = 0; i < 2; ++i)
#pragma unroll
        for (int r = 0; r < 8; ++r) {
          int row = r0 + 16 * i + hi * 8 + r, col = 16 * cj + l;
          Tout[(size_t)b * C_ * CI_ + (size_t)row * CI_ + col] = (f16)acc[i][r];
        }
    }
  }
}

// w_y[b][c][n] = T[b] @ theta[b] + W_b[c]; write f32 to d_out; accumulate BN sum/sumsq per channel.
// T_b (256x128 f16 = 64KB) is staged into LDS via the Tensor Data Mover (one D# per block).
__global__ void __launch_bounds__(256) k4_wy(const f16* __restrict__ Tbuf, const f16* __restrict__ th,
                                             const float* __restrict__ Wb, float* __restrict__ out,
                                             float* __restrict__ gsum, float* __restrict__ gssq) {
  __shared__ __align__(128) f16 Tlds[C_ * CI_];   // 64 KB (320 KB WGP LDS available)
  __shared__ float bsum[64], bssq[64];
  const int tid = threadIdx.x;
  if (tid < 64) { bsum[tid] = 0.f; bssq[tid] = 0.f; }
  const int lane = tid & 31, wid = tid >> 5;
  const int b = blockIdx.z;
  const f16* Tb = Tbuf + (size_t)b * C_ * CI_;

#if HAVE_TDM
  if (wid == 0) {
    // 2-D Tensor DMA descriptor: whole T_b tile, data_size=2B, dims 128 x 256, stride0=128.
    const unsigned long long ga = (unsigned long long)(uintptr_t)Tb;
    const unsigned int ldsoff = (unsigned int)(uintptr_t)&Tlds[0];  // LDS aperture: low 32 bits = byte offset
    u32x4 g0;
    g0[0] = 1u;                                               // count=1, user descriptor
    g0[1] = ldsoff;                                           // lds_addr
    g0[2] = (unsigned int)(ga & 0xFFFFFFFFu);                 // global_addr[31:0]
    g0[3] = (unsigned int)((ga >> 32) & 0x01FFFFFFu) | (2u << 30); // global_addr[56:32] | type=2
    i32x8 g1;
    g1[0] = 0x00010000;          // workgroup_mask=0 | data_size=1 (2B) | no flags
    g1[1] = (int)(128u << 16);   // tensor_dim0 = 128 (lo16)
    g1[2] = (int)(256u << 16);   // tensor_dim0 hi=0 | tensor_dim1 = 256 (lo16)
    g1[3] = (int)(128u << 16);   // tensor_dim1 hi=0 | tile_dim0 = 128
    g1[4] = 256;                 // tile_dim1 = 256 | tile_dim2 = 0
    g1[5] = 128;                 // tensor_dim0_stride = 128 (lo32)
    g1[6] = 0;
    g1[7] = 0;
    i32x4 gz = {0, 0, 0, 0};
#if defined(__clang_major__) && (__clang_major__ >= 23)
    i32x8 gz8 = {0, 0, 0, 0, 0, 0, 0, 0};
    __builtin_amdgcn_tensor_load_to_lds(g0, g1, gz, gz, gz8, 0);
#else
    __builtin_amdgcn_tensor_load_to_lds(g0, g1, gz, gz, 0);
#endif
    __builtin_amdgcn_s_wait_tensorcnt(0);
  }
#else
  // fallback: cooperative 16B copies
  for (int i = tid; i < (C_ * CI_) / 8; i += 256)
    ((uint4*)Tlds)[i] = ((const uint4*)Tb)[i];
#endif
  __syncthreads();

  const int wr = wid & 1, wc = wid >> 1;
  const int r0 = blockIdx.y * 64 + wr * 32;
  const int c0 = blockIdx.x * 128 + wc * 32;
  const f16* hb = th + (size_t)b * CI_ * N_;
  v8f acc[2][2] = {};
#pragma unroll
  for (int k0 = 0; k0 < CI_; k0 += 32) {
    v16h a0 = load_a16(Tlds, CI_, r0,      k0, lane);   // A from LDS (TDM-staged)
    v16h a1 = load_a16(Tlds, CI_, r0 + 16, k0, lane);
    v16h b0 = load_b16(hb, N_, k0, c0,      lane);
    v16h b1 = load_b16(hb, N_, k0, c0 + 16, lane);
    acc[0][0] = wmma16(a0, b0, acc[0][0]);
    acc[0][1] = wmma16(a0, b1, acc[0][1]);
    acc[1][0] = wmma16(a1, b0, acc[1][0]);
    acc[1][1] = wmma16(a1, b1, acc[1][1]);
  }
  const int l = lane & 15, hi = lane >> 4;
  float* ob = out + (size_t)b * C_ * N_;
#pragma unroll
  for (int i = 0; i < 2; ++i) {
#pragma unroll
    for (int r = 0; r < 8; ++r) {
      int row = r0 + 16 * i + hi * 8 + r;
      float wb = Wb[row];
      float v0 = acc[i][0][r] + wb;
      float v1 = acc[i][1][r] + wb;
      ob[(size_t)row * N_ + c0 + l]      = v0;
      ob[(size_t)row * N_ + c0 + 16 + l] = v1;
      float s = v0 + v1, q = v0 * v0 + v1 * v1;
#pragma unroll
      for (int msk = 1; msk <= 8; msk <<= 1) {   // reduce within 16-lane half (wave32)
        s += __shfl_xor(s, msk, 32);
        q += __shfl_xor(q, msk, 32);
      }
      if (l == 0) {
        int lrow = wr * 32 + 16 * i + hi * 8 + r;
        atomicAdd(&bsum[lrow], s);
        atomicAdd(&bssq[lrow], q);
      }
    }
  }
  __syncthreads();
  if (tid < 64) {
    atomicAdd(&gsum[blockIdx.y * 64 + tid], bsum[tid]);
    atomicAdd(&gssq[blockIdx.y * 64 + tid], bssq[tid]);
  }
}

// finalize: out = (w_y - mean) * rsqrt(var+eps) * gamma + beta + x  (in-place on d_out)
__global__ void k5_bn(const float* __restrict__ x, const float* __restrict__ gamma,
                      const float* __restrict__ beta, const float* __restrict__ gsum,
                      const float* __restrict__ gssq, float* __restrict__ out) {
  size_t idx = (size_t)blockIdx.x * 256 + threadIdx.x;
  int c = (int)((idx >> 12) & 255);
  const float inv = 1.0f / (float)(B_ * N_);
  float mean = gsum[c] * inv;
  float var  = gssq[c] * inv - mean * mean;
  float w = out[idx];
  out[idx] = (w - mean) * rsqrtf(var + 1e-5f) * gamma[c] + beta[c] + x[idx];
}

extern "C" void kernel_launch(void* const* d_in, const int* in_sizes, int n_in,
                              void* d_out, int out_size, void* d_ws, size_t ws_size,
                              hipStream_t stream) {
  (void)in_sizes; (void)n_in; (void)out_size; (void)ws_size;
  const float* x   = (const float*)d_in[0];
  const float* tw  = (const float*)d_in[1];
  const float* tb  = (const float*)d_in[2];
  const float* pw  = (const float*)d_in[3];
  const float* pb  = (const float*)d_in[4];
  const float* gw  = (const float*)d_in[5];
  const float* gb  = (const float*)d_in[6];
  const float* Ww  = (const float*)d_in[7];
  const float* Wb  = (const float*)d_in[8];
  const float* gam = (const float*)d_in[9];
  const float* bet = (const float*)d_in[10];
  float* out = (float*)d_out;
  char*  ws  = (char*)d_ws;
  f16*   wbf  = (f16*)(ws + OFF_WBF);
  f16*   th   = (f16*)(ws + OFF_TH);
  f16*   phiF = (f16*)(ws + OFF_PHIF);
  f16*   gF   = (f16*)(ws + OFF_GF);
  f16*   P    = (f16*)(ws + OFF_P);
  f16*   Gt   = (f16*)(ws + OFF_GT);
  f16*   Tbuf = (f16*)(ws + OFF_T);
  float* gsum = (float*)(ws + OFF_SUM);
  float* gssq = (float*)(ws + OFF_SSQ);

  k0_init<<<512, 256, 0, stream>>>(tw, pw, gw, Ww, wbf, gsum, gssq);

  dim3 gconv(32, 2, 16);
  k1_conv<<<gconv, 256, 0, stream>>>(x, wbf,         tb, th);
  k1_conv<<<gconv, 256, 0, stream>>>(x, wbf + 32768, pb, phiF);
  k1_conv<<<gconv, 256, 0, stream>>>(x, wbf + 65536, gb, gF);

  k2_pool<<<8192, 256, 0, stream>>>(phiF, P, 0);
  k2_pool<<<8192, 256, 0, stream>>>(gF,   Gt, 1);

  k3_ST<<<16, 256, 0, stream>>>(P, Gt, wbf + 98304, Tbuf);

  dim3 gwy(32, 4, 16);
  k4_wy<<<gwy, 256, 0, stream>>>(Tbuf, th, Wb, out, gsum, gssq);

  k5_bn<<<65536, 256, 0, stream>>>(x, gam, bet, gsum, gssq, out);
}
